// ModelGraph_39548058861639
// MI455X (gfx1250) — matrix operand order repaired and checked by reading, not compiled, not run on hardware
//
#include <hip/hip_runtime.h>
#include <hip/hip_bf16.h>

// Model dims
#define BB 32
#define SS 128
#define TT 64
#define VV 32000
#define EE 512
#define ENH 256
#define DEH 512

typedef __attribute__((ext_vector_type(16))) __bf16 v16bf;
typedef __attribute__((ext_vector_type(8)))  float  v8f;

#define DEV __device__ __forceinline__

union FragU { v16bf v; uint4 q[2]; };

// A-fragment (16x32, M x K): lane l (half=l>>4, l15=l&15) holds row M=l15,
// K = half*16 + {0..15} contiguous.
DEV v16bf load_a(const __bf16* base, long lda, int row, int k, int half) {
  FragU f;
  const __bf16* p = base + (long)row * lda + k + half * 16;
  f.q[0] = *(const uint4*)(p);
  f.q[1] = *(const uint4*)(p + 8);
  return f.v;
}
// B-fragment (32x16, K x N) from row-major W (N,K): lane holds column N=l15,
// K = half*16 + {0..15} contiguous.
DEV v16bf load_b(const __bf16* base, long ldw, int row, int k, int half) {
  FragU f;
  const __bf16* p = base + (long)row * ldw + k + half * 16;
  f.q[0] = *(const uint4*)(p);
  f.q[1] = *(const uint4*)(p + 8);
  return f.v;
}
DEV v8f wmma_bf(v16bf a, v16bf b, v8f c) {
  return __builtin_amdgcn_wmma_f32_16x16x32_bf16(false, a, false, b, (short)0, c,
                                                 false, false);
}
DEV float sigf(float x) { return 1.f / (1.f + expf(-x)); }

// ---------------------------------------------------------------------------
// bf16 WMMA GEMM: C[M,N] = A[M,K] @ W[N,K]^T + bias, f32 and/or bf16 out.
// Block = 8 waves sharing one m-tile: the 16xK A panel is staged in LDS once
// per block (A fragments then come from ds_load_b128), and the per-wave B
// fragment stream is double-buffered so each WMMA overlaps the next loads.
// grid = ( ceil(N/16 / 8), M/16 ).
// ---------------------------------------------------------------------------
__global__ void __launch_bounds__(256) wmma_gemm(
    const __bf16* __restrict__ A, int lda,
    const __bf16* __restrict__ W, int ldw,
    const float* __restrict__ bias,
    float* __restrict__ C, long ldc,
    __bf16* __restrict__ Cbf, long ldcbf,
    int N, int K) {
  __shared__ __bf16 aPanel[16 * 1024];  // K <= 1024
  int mT = blockIdx.y;
  // --- cooperative stage of A panel (16 rows x K) into LDS, 16B chunks ---
  int chunksPerRow = K >> 3;  // uint4 = 8 bf16
  int totalChunks = 16 * chunksPerRow;
  for (int idx = threadIdx.x; idx < totalChunks; idx += 256) {
    int row = idx / chunksPerRow;
    int ck = idx - row * chunksPerRow;
    *(uint4*)&aPanel[row * K + ck * 8] =
        *(const uint4*)(A + (long)(mT * 16 + row) * lda + ck * 8);
  }
  __syncthreads();

  int wave = threadIdx.x >> 5;
  int lane = threadIdx.x & 31;
  int nTiles = N >> 4;
  int nT = blockIdx.x * 8 + wave;
  if (nT < nTiles) {
    int half = lane >> 4, l15 = lane & 15;
    int brow = nT * 16 + l15;
    const __bf16* wrow = W + (long)brow * ldw;
    v8f acc = {};
    // software pipeline: b for step k is loaded one iteration ahead
    v16bf b_cur;
    {
      FragU f;
      const __bf16* p = wrow + half * 16;
      f.q[0] = *(const uint4*)(p);
      f.q[1] = *(const uint4*)(p + 8);
      b_cur = f.v;
    }
    for (int k = 0; k < K; k += 32) {
      v16bf b_nxt = b_cur;
      if (k + 32 < K) {
        __builtin_prefetch(wrow + k + 64, 0, 1);
        FragU f;
        const __bf16* p = wrow + k + 32 + half * 16;
        f.q[0] = *(const uint4*)(p);
        f.q[1] = *(const uint4*)(p + 8);
        b_nxt = f.v;
      }
      v16bf a = load_a(aPanel, K, l15, k, half);  // ds_load_b128
      acc = wmma_bf(a, b_cur, acc);
      b_cur = b_nxt;
    }
    int col = nT * 16 + l15;
    float bv = bias ? bias[col] : 0.f;
    for (int r = 0; r < 8; ++r) {
      int m = mT * 16 + r + half * 8;
      float v = acc[r] + bv;
      if (C)   C[(long)m * ldc + col] = v;
      if (Cbf) Cbf[(long)m * ldcbf + col] = (__bf16)v;
    }
  }
}

// ---------------------------------------------------------------------------
// Persistent LSTM scan (one workgroup = 32 waves). Input-side GEMM (Gx,
// bias folded) is precomputed; here only h @ whh^T per step via WMMA with
// h kept as bf16 in LDS (A-fragments via ds_load_b128). Wave (mT, jt) owns
// the i/f/g/o tiles for rows mT*16..+15, cols jt*16..+15 -> pointwise LSTM
// update fully in-register. Next step's Gx slice is prefetched during WMMA.
// ---------------------------------------------------------------------------
__global__ void __launch_bounds__(1024) lstm_scan(
    const float* __restrict__ Gx,    // (S, 32, 1024) = x@wih^T + b
    const __bf16* __restrict__ Whh,  // (1024, 256)
    __bf16* __restrict__ Ybf, long ySS, long yBS,
    float* __restrict__ Yf, long fSS, long fBS,
    __bf16* __restrict__ hFinal, int hFinalLd,
    int steps, int rev) {
  __shared__ __bf16 h_sh[BB * ENH];
  __shared__ float  c_sh[BB * ENH];
  for (int i = threadIdx.x; i < BB * ENH; i += blockDim.x) {
    h_sh[i] = (__bf16)0.f;
    c_sh[i] = 0.f;
  }
  __syncthreads();
  int wave = threadIdx.x >> 5, lane = threadIdx.x & 31;
  int mT = wave >> 4, jt = wave & 15;
  int half = lane >> 4, l15 = lane & 15;
  int arow = mT * 16 + l15;
  int j = jt * 16 + l15;
  for (int t = 0; t < steps; ++t) {
    int s = rev ? (steps - 1 - t) : t;
    if (t + 1 < steps) {
      int sn = rev ? (steps - 2 - t) : (t + 1);
      // each thread touches one cache line of next step's Gx slice (128KB)
      __builtin_prefetch(Gx + (long)sn * BB * 1024 + (long)threadIdx.x * 32, 0, 1);
    }
    v8f a0 = {}, a1 = {}, a2 = {}, a3 = {};
    for (int k = 0; k < ENH; k += 32) {
      v16bf a = load_a(h_sh, ENH, arow, k, half);
      v16bf b0 = load_b(Whh, ENH, 0 * ENH + jt * 16 + l15, k, half);
      v16bf b1 = load_b(Whh, ENH, 1 * ENH + jt * 16 + l15, k, half);
      v16bf b2 = load_b(Whh, ENH, 2 * ENH + jt * 16 + l15, k, half);
      v16bf b3 = load_b(Whh, ENH, 3 * ENH + jt * 16 + l15, k, half);
      a0 = wmma_bf(a, b0, a0);
      a1 = wmma_bf(a, b1, a1);
      a2 = wmma_bf(a, b2, a2);
      a3 = wmma_bf(a, b3, a3);
    }
    const float* gxs = Gx + (long)s * BB * 1024;
    float hnew[8];
    for (int r = 0; r < 8; ++r) {
      int m = mT * 16 + r + half * 8;
      float gi = a0[r] + gxs[m * 1024 + 0 * ENH + j];
      float gf = a1[r] + gxs[m * 1024 + 1 * ENH + j];
      float gg = a2[r] + gxs[m * 1024 + 2 * ENH + j];
      float go = a3[r] + gxs[m * 1024 + 3 * ENH + j];
      float cn = sigf(gf) * c_sh[m * ENH + j] + sigf(gi) * tanhf(gg);
      c_sh[m * ENH + j] = cn;  // element owned exclusively by this lane
      hnew[r] = sigf(go) * tanhf(cn);
    }
    __syncthreads();  // all waves done reading h_sh for this step
    for (int r = 0; r < 8; ++r) {
      int m = mT * 16 + r + half * 8;
      __bf16 hv = (__bf16)hnew[r];
      h_sh[m * ENH + j] = hv;
      Ybf[(long)s * ySS + (long)m * yBS + j] = hv;
      if (Yf) Yf[(long)s * fSS + (long)m * fBS + j] = hnew[r];
    }
    __syncthreads();
  }
  for (int r = 0; r < 8; ++r) {
    int m = mT * 16 + r + half * 8;
    hFinal[(long)m * hFinalLd + j] = h_sh[m * ENH + j];
  }
}

// ---------------------------------------------------------------------------
// Decoder LSTM cell: gates = di@d_wih^T + h@d_whh^T + b, N=2048, K=512+512.
// Grid = 2 blocks (mT) x 32 waves (jt over DE/16). h read from hid_bf[:,0:512],
// c from f32 buffer; writes new h|c (bf16) into hid_bf and c (f32).
// ---------------------------------------------------------------------------
__global__ void __launch_bounds__(1024) dec_cell(
    const __bf16* __restrict__ di_bf,  // (32,512)
    __bf16* __restrict__ hid_bf,       // (32,1024) [h | c]
    float* __restrict__ c_f32,         // (32,512)
    const __bf16* __restrict__ Wih,    // (2048,512)
    const __bf16* __restrict__ Whh,    // (2048,512)
    const float* __restrict__ bias) {  // (2048)
  int mT = blockIdx.x;
  int wave = threadIdx.x >> 5, lane = threadIdx.x & 31;
  int jt = wave;
  int half = lane >> 4, l15 = lane & 15;
  int arow = mT * 16 + l15;
  int j = jt * 16 + l15;
  v8f acc[4] = {{}, {}, {}, {}};
  for (int k = 0; k < DEH; k += 32) {
    v16bf a = load_a(di_bf, DEH, arow, k, half);
    for (int g = 0; g < 4; ++g) {
      v16bf b = load_b(Wih, DEH, g * DEH + jt * 16 + l15, k, half);
      acc[g] = wmma_bf(a, b, acc[g]);
    }
  }
  for (int k = 0; k < DEH; k += 32) {
    v16bf a = load_a(hid_bf, 1024, arow, k, half);  // old h
    for (int g = 0; g < 4; ++g) {
      v16bf b = load_b(Whh, DEH, g * DEH + jt * 16 + l15, k, half);
      acc[g] = wmma_bf(a, b, acc[g]);
    }
  }
  float hn[8], cn[8];
  for (int r = 0; r < 8; ++r) {
    int m = mT * 16 + r + half * 8;
    float gi = acc[0][r] + bias[0 * DEH + j];
    float gf = acc[1][r] + bias[1 * DEH + j];
    float gg = acc[2][r] + bias[2 * DEH + j];
    float go = acc[3][r] + bias[3 * DEH + j];
    cn[r] = sigf(gf) * c_f32[m * DEH + j] + sigf(gi) * tanhf(gg);
    hn[r] = sigf(go) * tanhf(cn[r]);
  }
  __syncthreads();  // all waves in this block done reading old h
  for (int r = 0; r < 8; ++r) {
    int m = mT * 16 + r + half * 8;
    hid_bf[m * 1024 + j] = (__bf16)hn[r];
    hid_bf[m * 1024 + DEH + j] = (__bf16)cn[r];
    c_f32[m * DEH + j] = cn[r];
  }
}

// ---------------------------------------------------------------------------
// Fused attention step: scores = tanh(en_proj + hidproj) . v (masked),
// softmax over S, ctx = aw @ en_out; writes [e_t | ctx] as bf16.
// One block per batch row, 128 threads (one per s).
// ---------------------------------------------------------------------------
__global__ void __launch_bounds__(128) attn_step(
    const float* __restrict__ en_proj,  // (B,S,DE)
    const float* __restrict__ hidproj,  // (B,DE) incl. attnb
    const float* __restrict__ attn_v,   // (DE)
    const int* __restrict__ x_len,      // (B)
    const float* __restrict__ en_out,   // (B,S,512)
    const __bf16* __restrict__ e_t,     // (B,512)
    __bf16* __restrict__ di_cat) {      // (B,1024)
  int b = blockIdx.x, tid = threadIdx.x;
  __shared__ float sc[SS];
  __shared__ float red[SS];
  const float* ep = en_proj + ((long)b * SS + tid) * DEH;
  const float* hp = hidproj + (long)b * DEH;
  float acc = 0.f;
  for (int d = 0; d < DEH; ++d) acc += tanhf(ep[d] + hp[d]) * attn_v[d];
  bool valid = tid < x_len[b];
  red[tid] = valid ? acc : -__builtin_inff();
  __syncthreads();
  for (int off = 64; off > 0; off >>= 1) {
    if (tid < off) red[tid] = fmaxf(red[tid], red[tid + off]);
    __syncthreads();
  }
  float mx = red[0];
  __syncthreads();
  float e = valid ? expf(acc - mx) : 0.f;
  red[tid] = e;
  __syncthreads();
  for (int off = 64; off > 0; off >>= 1) {
    if (tid < off) red[tid] += red[tid + off];
    __syncthreads();
  }
  float inv = 1.f / red[0];
  __syncthreads();
  sc[tid] = e * inv;
  __syncthreads();
  for (int d = tid; d < DEH; d += 128) {
    float c = 0.f;
    const float* eo = en_out + (long)b * SS * DEH + d;
    for (int s = 0; s < SS; ++s) c += sc[s] * eo[(long)s * DEH];
    di_cat[(long)b * 1024 + DEH + d] = (__bf16)c;
    di_cat[(long)b * 1024 + d] = e_t[(long)b * EE + d];
  }
}

// ---------------------------------------------------------------------------
// log_softmax over V=32000 and scatter into (B,T,V) output. Block per row.
// ---------------------------------------------------------------------------
__global__ void __launch_bounds__(1024) logsoftmax_out(
    const float* __restrict__ logits,  // (32, V)
    float* __restrict__ out, int t) {
  int b = blockIdx.x, tid = threadIdx.x;
  __shared__ float red[1024];
  const float* lg = logits + (long)b * VV;
  float mx = -__builtin_inff();
  for (int v = tid; v < VV; v += 1024) mx = fmaxf(mx, lg[v]);
  red[tid] = mx;
  __syncthreads();
  for (int off = 512; off > 0; off >>= 1) {
    if (tid < off) red[tid] = fmaxf(red[tid], red[tid + off]);
    __syncthreads();
  }
  mx = red[0];
  __syncthreads();
  float sum = 0.f;
  for (int v = tid; v < VV; v += 1024) sum += expf(lg[v] - mx);
  red[tid] = sum;
  __syncthreads();
  for (int off = 512; off > 0; off >>= 1) {
    if (tid < off) red[tid] += red[tid + off];
    __syncthreads();
  }
  float lz = mx + logf(red[0]);
  float* ob = out + ((long)b * TT + t) * VV;
  for (int v = tid; v < VV; v += 1024) ob[v] = lg[v] - lz;
}

// ---------------------------------------------------------------------------
// Small utility kernels
// ---------------------------------------------------------------------------
__global__ void cast_bf(const float* __restrict__ src, __bf16* __restrict__ dst,
                        long n) {
  long i = (long)blockIdx.x * blockDim.x + threadIdx.x;
  if (i < n) dst[i] = (__bf16)src[i];
}

__global__ void gather_enc(const float* __restrict__ emb,
                           const int* __restrict__ x,
                           __bf16* __restrict__ out) {  // (S,B,E)
  long i = (long)blockIdx.x * blockDim.x + threadIdx.x;
  if (i >= (long)SS * BB * EE) return;
  int e = (int)(i & (EE - 1));
  long r = i >> 9;
  int b = (int)(r & (BB - 1));
  int s = (int)(r >> 5);
  int tok = x[b * SS + s];
  out[i] = (__bf16)emb[(long)tok * EE + e];
}

__global__ void gather_dec(const float* __restrict__ emb,
                           const int* __restrict__ y_in,
                           const int* __restrict__ y_out,
                           __bf16* __restrict__ out) {  // (T,B,E)
  long i = (long)blockIdx.x * blockDim.x + threadIdx.x;
  if (i >= (long)TT * BB * EE) return;
  int e = (int)(i & (EE - 1));
  long r = i >> 9;
  int b = (int)(r & (BB - 1));
  int t = (int)(r >> 5);
  int tok = (t == 0) ? y_in[b * TT] : y_out[b * TT + (t - 1)];
  out[i] = (__bf16)emb[(long)tok * EE + e];
}

// ---------------------------------------------------------------------------
// Host orchestration
// ---------------------------------------------------------------------------
extern "C" void kernel_launch(void* const* d_in, const int* in_sizes, int n_in,
                              void* d_out, int out_size, void* d_ws,
                              size_t ws_size, hipStream_t stream) {
  (void)in_sizes; (void)n_in; (void)out_size; (void)ws_size;

  const int*   x        = (const int*)d_in[0];
  const int*   x_len    = (const int*)d_in[1];
  const int*   y_in     = (const int*)d_in[2];
  const int*   y_out    = (const int*)d_in[3];
  const float* enc_emb  = (const float*)d_in[5];
  const float* e0f_wih  = (const float*)d_in[6];
  const float* e0f_whh  = (const float*)d_in[7];
  const float* e0f_b    = (const float*)d_in[8];
  const float* e0b_wih  = (const float*)d_in[9];
  const float* e0b_whh  = (const float*)d_in[10];
  const float* e0b_b    = (const float*)d_in[11];
  const float* e1f_wih  = (const float*)d_in[12];
  const float* e1f_whh  = (const float*)d_in[13];
  const float* e1f_b    = (const float*)d_in[14];
  const float* e1b_wih  = (const float*)d_in[15];
  const float* e1b_whh  = (const float*)d_in[16];
  const float* e1b_b    = (const float*)d_in[17];
  const float* hW       = (const float*)d_in[18];
  const float* hb       = (const float*)d_in[19];
  const float* cW       = (const float*)d_in[20];
  const float* cb       = (const float*)d_in[21];
  const float* dec_emb  = (const float*)d_in[22];
  const float* attnW    = (const float*)d_in[23];
  const float* attnb    = (const float*)d_in[24];
  const float* attn_v   = (const float*)d_in[25];
  const float* combW    = (const float*)d_in[26];
  const float* combb    = (const float*)d_in[27];
  const float* d_wih    = (const float*)d_in[28];
  const float* d_whh    = (const float*)d_in[29];
  const float* d_b      = (const float*)d_in[30];
  const float* outW     = (const float*)d_in[31];
  const float* outb     = (const float*)d_in[32];
  float* out = (float*)d_out;

  char* ws = (char*)d_ws;
  size_t off = 0;
  auto alloc = [&](size_t bytes) -> void* {
    void* p = ws + off;
    off = (off + bytes + 255) & ~(size_t)255;
    return p;
  };

  // bf16 weight copies
  __bf16* w_e0f_wih = (__bf16*)alloc(2L * 1024 * 512);
  __bf16* w_e0f_whh = (__bf16*)alloc(2L * 1024 * 256);
  __bf16* w_e0b_wih = (__bf16*)alloc(2L * 1024 * 512);
  __bf16* w_e0b_whh = (__bf16*)alloc(2L * 1024 * 256);
  __bf16* w_e1f_wih = (__bf16*)alloc(2L * 1024 * 512);
  __bf16* w_e1f_whh = (__bf16*)alloc(2L * 1024 * 256);
  __bf16* w_e1b_wih = (__bf16*)alloc(2L * 1024 * 512);
  __bf16* w_e1b_whh = (__bf16*)alloc(2L * 1024 * 256);
  __bf16* w_hW      = (__bf16*)alloc(2L * 512 * 1024);
  __bf16* w_cW      = (__bf16*)alloc(2L * 512 * 1024);
  __bf16* w_attnW   = (__bf16*)alloc(2L * 512 * 1536);
  __bf16* w_combW   = (__bf16*)alloc(2L * 512 * 1024);
  __bf16* w_dwih    = (__bf16*)alloc(2L * 2048 * 512);
  __bf16* w_dwhh    = (__bf16*)alloc(2L * 2048 * 512);
  __bf16* w_outW    = (__bf16*)alloc(2L * VV * 512);
  // activations
  __bf16* emb_bf    = (__bf16*)alloc(2L * SS * BB * EE);
  float*  GxA       = (float*)alloc(4L * SS * BB * 1024);
  float*  GxB       = (float*)alloc(4L * SS * BB * 1024);
  __bf16* x1_bf     = (__bf16*)alloc(2L * SS * BB * 512);
  float*  en_out_f  = (float*)alloc(4L * BB * SS * 512);
  __bf16* en_out_bf = (__bf16*)alloc(2L * BB * SS * 512);
  float*  en_proj   = (float*)alloc(4L * BB * SS * DEH);
  __bf16* hcat_bf   = (__bf16*)alloc(2L * BB * 1024);
  __bf16* hid_bf    = (__bf16*)alloc(2L * BB * 1024);
  float*  dec_c     = (float*)alloc(4L * BB * DEH);
  float*  hidproj   = (float*)alloc(4L * BB * DEH);
  __bf16* di_cat    = (__bf16*)alloc(2L * BB * 1024);
  __bf16* di_bf     = (__bf16*)alloc(2L * BB * DEH);
  __bf16* emb_t     = (__bf16*)alloc(2L * TT * BB * EE);
  float*  logits    = (float*)alloc(4L * BB * VV);

  auto cast = [&](const float* s, __bf16* d, long n) {
    cast_bf<<<(int)((n + 255) / 256), 256, 0, stream>>>(s, d, n);
  };
  auto gemm = [&](const __bf16* A, int lda, const __bf16* W, int ldw,
                  const float* bias, float* C, long ldc, __bf16* Cbf,
                  long ldcbf, int M, int N, int K) {
    dim3 grid((N / 16 + 7) / 8, M / 16);
    wmma_gemm<<<grid, 256, 0, stream>>>(A, lda, W, ldw, bias, C, ldc, Cbf,
                                        ldcbf, N, K);
  };

  // --- weight casts ---
  cast(e0f_wih, w_e0f_wih, 1024L * 512);
  cast(e0f_whh, w_e0f_whh, 1024L * 256);
  cast(e0b_wih, w_e0b_wih, 1024L * 512);
  cast(e0b_whh, w_e0b_whh, 1024L * 256);
  cast(e1f_wih, w_e1f_wih, 1024L * 512);
  cast(e1f_whh, w_e1f_whh, 1024L * 256);
  cast(e1b_wih, w_e1b_wih, 1024L * 512);
  cast(e1b_whh, w_e1b_whh, 1024L * 256);
  cast(hW, w_hW, 512L * 1024);
  cast(cW, w_cW, 512L * 1024);
  cast(attnW, w_attnW, 512L * 1536);
  cast(combW, w_combW, 512L * 1024);
  cast(d_wih, w_dwih, 2048L * 512);
  cast(d_whh, w_dwhh, 2048L * 512);
  cast(outW, w_outW, (long)VV * 512);

  // --- embeddings ---
  gather_enc<<<(SS * BB * EE + 255) / 256, 256, 0, stream>>>(enc_emb, x, emb_bf);
  gather_dec<<<(TT * BB * EE + 255) / 256, 256, 0, stream>>>(dec_emb, y_in,
                                                             y_out, emb_t);

  // --- encoder layer 0: hoisted input GEMMs (bias folded), then scans ---
  gemm(emb_bf, EE, w_e0f_wih, EE, e0f_b, GxA, 1024, nullptr, 0, SS * BB, 1024, EE);
  gemm(emb_bf, EE, w_e0b_wih, EE, e0b_b, GxB, 1024, nullptr, 0, SS * BB, 1024, EE);
  // x1 layout (S,B,512): sStride = B*512, bStride = 512
  lstm_scan<<<1, 1024, 0, stream>>>(GxA, w_e0f_whh, x1_bf + 0, (long)BB * 512,
                                    512, nullptr, 0, 0, hcat_bf + 0, 1024, SS, 0);
  lstm_scan<<<1, 1024, 0, stream>>>(GxB, w_e0b_whh, x1_bf + 256, (long)BB * 512,
                                    512, nullptr, 0, 0, hcat_bf + 256, 1024, SS, 1);

  // --- encoder layer 1 ---
  gemm(x1_bf, 512, w_e1f_wih, 512, e1f_b, GxA, 1024, nullptr, 0, SS * BB, 1024, 512);
  gemm(x1_bf, 512, w_e1b_wih, 512, e1b_b, GxB, 1024, nullptr, 0, SS * BB, 1024, 512);
  // en_out layout (B,S,512): sStride = 512, bStride = S*512
  lstm_scan<<<1, 1024, 0, stream>>>(GxA, w_e1f_whh, en_out_bf + 0, 512,
                                    (long)SS * 512, en_out_f + 0, 512,
                                    (long)SS * 512, hcat_bf + 512, 1024, SS, 0);
  lstm_scan<<<1, 1024, 0, stream>>>(GxB, w_e1b_whh, en_out_bf + 256, 512,
                                    (long)SS * 512, en_out_f + 256, 512,
                                    (long)SS * 512, hcat_bf + 768, 1024, SS, 1);

  // --- decoder init: h0 -> hid_bf[:, :512], c0 -> dec_c + hid_bf[:, 512:] ---
  gemm(hcat_bf, 1024, w_hW, 1024, hb, nullptr, 0, hid_bf, 1024, BB, DEH, 1024);
  gemm(hcat_bf, 1024, w_cW, 1024, cb, dec_c, DEH, hid_bf + DEH, 1024, BB, DEH, 1024);

  // --- attention pre-projection: en_proj = en_out @ We^T  (We = attnW[:,1024:]) ---
  gemm(en_out_bf, 512, w_attnW + 2 * DEH, 1536, nullptr, en_proj, DEH, nullptr,
       0, BB * SS, DEH, 512);

  // --- decoder loop ---
  for (int t = 0; t < TT; ++t) {
    // hidproj = [h|c] @ Wh^T + attnb   (Wh = attnW[:, :1024])
    gemm(hid_bf, 1024, w_attnW, 1536, attnb, hidproj, DEH, nullptr, 0, BB, DEH,
         1024);
    attn_step<<<BB, SS, 0, stream>>>(en_proj, hidproj, attn_v, x_len, en_out_f,
                                     emb_t + (long)t * BB * EE, di_cat);
    // di = [e_t|ctx] @ combW^T + combb  -> bf16
    gemm(di_cat, 1024, w_combW, 1024, combb, nullptr, 0, di_bf, DEH, BB, DEH,
         1024);
    dec_cell<<<2, 1024, 0, stream>>>(di_bf, hid_bf, dec_c, w_dwih, w_dwhh, d_b);
    // logits = h2 @ outW^T + outb   (dominant GEMM: M=32, N=32000, K=512)
    gemm(hid_bf, 1024, w_outW, 512, outb, logits, VV, nullptr, 0, BB, VV, 512);
    logsoftmax_out<<<BB, 1024, 0, stream>>>(logits, out, t);
  }
}
